// TFNO3DModule_24507083391134
// MI455X (gfx1250) — matrix operand order repaired and checked by reading, not compiled, not run on hardware
//
#include <hip/hip_runtime.h>
#include <hip/hip_bf16.h>
#include <cmath>

typedef __bf16 bf16_t;
typedef __attribute__((ext_vector_type(16))) __bf16 v16bf;
typedef __attribute__((ext_vector_type(8)))  float  v8f;

#define PI2   6.28318530717958647692f
#define PLANE 16384L      // H*W
#define VOL   163840L     // T_IN*H*W
#define CH    64
#define MODES 4352L       // 8*32*17

// gelu(x) ~= 0.5x(1+tanh(u)) = x * sigmoid(2u),  u = 0.79788456*(x + 0.044715 x^3)
// branch-free: v_exp_f32 + v_rcp_f32 + mul; clamp keeps e+1 a normal positive float.
__device__ __forceinline__ float gelu_tanh(float x){
  float u = 1.5957691216057308f * (x + 0.044715f * x * x * x);  // 2u
  u = fminf(fmaxf(u, -30.0f), 30.0f);
  float e = __expf(u);
  return x * (e * __builtin_amdgcn_rcpf(e + 1.0f));
}

// ---------------- WMMA helpers (gfx1250 wave32, 16x16x32 bf16) ---------------------
__device__ __forceinline__ v8f wmma_bf16(v16bf a, v16bf b, v8f c){
  return __builtin_amdgcn_wmma_f32_16x16x32_bf16(false, a, false, b, (short)0, c, false, false);
}

// B (32x16): lane = 16g+n ; K block = 16g + 2j (+1).  Branch-free: clamp addr + select 0.
template<int KMAX, long CS>
__device__ __forceinline__ v16bf load_bfrag(const float* __restrict__ in, int pofs, int k0, int lane){
  int g = (lane >> 4) & 1;
  int n = lane & 15;
  const float* col = in + pofs + n;
  v16bf b;
  #pragma unroll
  for (int j = 0; j < 8; ++j){
    int k  = k0 + 16*g + 2*j;
    int c0 = (k     < KMAX) ? k     : 0;   // always-safe address
    int c1 = ((k+1) < KMAX) ? (k+1) : 0;
    float x0 = col[(long)c0*CS];
    float x1 = col[(long)c1*CS];
    x0 = (k     < KMAX) ? x0 : 0.0f;       // v_cndmask, no exec branch
    x1 = ((k+1) < KMAX) ? x1 : 0.0f;
    b[2*j]   = (bf16_t)x0;
    b[2*j+1] = (bf16_t)x1;
  }
  return b;
}

// A fragments are pre-packed to fragment order: one aligned v16bf (32B) per lane.
__device__ __forceinline__ v16bf load_afrag_packed(const v16bf* __restrict__ pA,
                                                   int mt, int ks, int KS, int lane){
  return pA[(mt*KS + ks)*32 + lane];
}

// ---------------- weight repack: row-major f32 [M x K] -> fragment-ordered bf16 ----
// A (16x32): lane = 16g+m ; element e -> K = 16g+e
__global__ void __launch_bounds__(256)
repack_a_kernel(const float* __restrict__ W, int M, int K, bf16_t* __restrict__ out, int total){
  int idx = blockIdx.x*256 + threadIdx.x;      // one thread per (tile-kstep, lane)
  if (idx >= total) return;
  int lane = idx & 31, tk = idx >> 5;
  int KS = (K + 31) >> 5;
  int g = (lane >> 4) & 1, m = lane & 15;
  int mt = tk / KS, ks = tk % KS;
  int row = mt*16 + m;
  bf16_t* dst = out + (size_t)tk*32*16 + (size_t)lane*16;
  #pragma unroll
  for (int e = 0; e < 16; ++e){
    int kb = 16*g + e;
    int k  = ks*32 + kb;
    float val = (row < M && k < K) ? W[(long)row*K + k] : 0.0f;
    dst[e] = (bf16_t)val;
  }
}

// ---------------- head copy: out[:, :15] = x[:, :15] -------------------------------
__global__ void __launch_bounds__(256)
copy_head_kernel(const float4* __restrict__ x, float4* __restrict__ out){
  long i = (long)blockIdx.x*256 + threadIdx.x;       // 1,966,080 float4
  long b = i / 491520, r = i % 491520;               // 15*8*16384/4 per batch
  out[b*819200 + r] = x[b*819200 + r];               // 25*8*16384/4 batch stride
}

// ---------------- fused 2-layer MLP (lift / proj) via WMMA + LDS -------------------
// MODE 0: lift  8->256(gelu)->64 ; input x (B,25,8,H,W) window t<10 ; out v (B,64,T,H,W)
// MODE 1: proj 64->256(gelu)->8  ; input v ; out d_out frames 15..24
template<int MODE>
__global__ void __launch_bounds__(256)
mlp_wmma_kernel(const float* __restrict__ in, const v16bf* __restrict__ pW1,
                const float* __restrict__ b1, const v16bf* __restrict__ pW2,
                const float* __restrict__ b2, float* __restrict__ out){
  __shared__ __attribute__((aligned(32))) bf16_t hid[64*272];  // [n][k] k-contig, padded
  const int tid = threadIdx.x;
  const int wave = tid >> 5, lane = tid & 31;
  const int g = lane >> 4, l = lane & 15;

  long ptile = (long)blockIdx.x * 64;
  int  b   = (int)(ptile / VOL);
  long p2  = ptile % VOL;
  int  t   = (int)(p2 >> 14);
  long hwb = p2 & 16383;

  const float* sbase = (MODE==0) ? (in + ((long)(b*25 + t)*8)*PLANE + hwb)
                                 : (in + (long)b*CH*VOL + p2);

  const v8f vz = {0.f,0.f,0.f,0.f,0.f,0.f,0.f,0.f};

  // ---- stage 1: hidden(256) = gelu(W1*in + b1); wave owns mtiles {2w,2w+1} x nt 0..3
  v8f acc[2][4];
  #pragma unroll
  for (int mi=0; mi<2; ++mi)
    #pragma unroll
    for (int nt=0; nt<4; ++nt) acc[mi][nt] = vz;

  constexpr int KS1 = (MODE==0) ? 1 : 2;
  #pragma unroll
  for (int ks = 0; ks < KS1; ++ks){
    int k0 = ks*32;
    v16bf bf[4];
    #pragma unroll
    for (int nt = 0; nt < 4; ++nt){
      if (MODE==0) bf[nt] = load_bfrag<8,  PLANE>(sbase, nt*16, k0, lane);
      else         bf[nt] = load_bfrag<64, VOL  >(sbase, nt*16, k0, lane);
    }
    #pragma unroll
    for (int mi = 0; mi < 2; ++mi){
      v16bf a = load_afrag_packed(pW1, 2*wave+mi, ks, KS1, lane);
      #pragma unroll
      for (int nt = 0; nt < 4; ++nt)
        acc[mi][nt] = wmma_bf16(a, bf[nt], acc[mi][nt]);
    }
  }
  #pragma unroll
  for (int mi = 0; mi < 2; ++mi){
    #pragma unroll
    for (int nt = 0; nt < 4; ++nt){
      int n = nt*16 + l;
      #pragma unroll
      for (int r = 0; r < 8; ++r){
        int k = (2*wave+mi)*16 + 8*g + r;
        hid[n*272 + k] = (bf16_t)gelu_tanh(acc[mi][nt][r] + b1[k]);
      }
    }
  }
  __syncthreads();

  // ---- stage 2: out = W2*hidden + b2 (K=256 from LDS) ----
  if (MODE==0){
    int mt = wave >> 1, nbase = (wave & 1)*2;
    v8f a0 = vz, a1 = vz;
    #pragma unroll
    for (int ks = 0; ks < 8; ++ks){
      int k0 = ks*32;
      v16bf a  = load_afrag_packed(pW2, mt, ks, 8, lane);
      v16bf f0 = *(const v16bf*)&hid[((nbase+0)*16 + l)*272 + k0 + 16*g];
      v16bf f1 = *(const v16bf*)&hid[((nbase+1)*16 + l)*272 + k0 + 16*g];
      a0 = wmma_bf16(a, f0, a0);
      a1 = wmma_bf16(a, f1, a1);
    }
    #pragma unroll
    for (int e = 0; e < 2; ++e){
      v8f av = e ? a1 : a0;
      long n = (nbase+e)*16 + l;
      #pragma unroll
      for (int r = 0; r < 8; ++r){
        int o = mt*16 + 8*g + r;
        out[(long)(b*CH + o)*VOL + p2 + n] = av[r] + b2[o];
      }
    }
  } else {
    if (wave < 4){
      int nt = wave;
      v8f ap = vz;
      #pragma unroll
      for (int ks = 0; ks < 8; ++ks){
        int k0 = ks*32;
        v16bf a  = load_afrag_packed(pW2, 0, ks, 8, lane);
        v16bf f0 = *(const v16bf*)&hid[(nt*16 + l)*272 + k0 + 16*g];
        ap = wmma_bf16(a, f0, ap);
      }
      if (g == 0){
        #pragma unroll
        for (int r = 0; r < 8; ++r){
          int o = r;  // valid rows only (COUT=8)
          out[((long)(b*25 + 15 + t)*8 + o)*PLANE + hwb + nt*16 + l] = ap[r] + b2[o];
        }
      }
    }
  }
}

// ---------------- skip (64x64) WMMA fused with spectral residual + GELU ------------
__global__ void __launch_bounds__(256)
skip_wmma_kernel(const float* __restrict__ vin, const float* __restrict__ S,
                 const v16bf* __restrict__ pWsk, const float* __restrict__ bsk,
                 float* __restrict__ vout, int act){
  const int tid = threadIdx.x;
  const int wave = tid >> 5, lane = tid & 31;
  long ptile = (long)blockIdx.x * 64;
  int  b  = (int)(ptile / VOL);
  long p2 = ptile % VOL;
  const float* inb = vin + (long)b*CH*VOL + p2;
  const float* Sbp = S   + (long)b*CH*VOL + p2;
  float*       ob  = vout + (long)b*CH*VOL + p2;
  int mt = wave >> 1;
  int nbase = (wave & 1) * 2;
  const v8f vz = {0.f,0.f,0.f,0.f,0.f,0.f,0.f,0.f};
  v8f a0 = vz, a1 = vz;
  #pragma unroll
  for (int ks = 0; ks < 2; ++ks){
    int k0 = ks*32;
    v16bf a   = load_afrag_packed(pWsk, mt, ks, 2, lane);
    v16bf bf0 = load_bfrag<64, VOL>(inb, (nbase+0)*16, k0, lane);
    v16bf bf1 = load_bfrag<64, VOL>(inb, (nbase+1)*16, k0, lane);
    a0 = wmma_bf16(a, bf0, a0);
    a1 = wmma_bf16(a, bf1, a1);
  }
  int g = lane >> 4, l = lane & 15;
  #pragma unroll
  for (int e = 0; e < 2; ++e){
    v8f av = e ? a1 : a0;
    long n = (nbase+e)*16 + l;
    #pragma unroll
    for (int r = 0; r < 8; ++r){
      int o = mt*16 + 8*g + r;
      float val = av[r] + bsk[o] + Sbp[(long)o*VOL + n];
      if (act) val = gelu_tanh(val);
      ob[(long)o*VOL + n] = val;
    }
  }
}

// ---------------- truncated DFT stages (norm="forward" folded into F1) -------------
// F1: rfft over W, keep kw<17.  v (B,C,T,H,W) -> f1 (B,C,T,KW17,H) complex
__global__ void __launch_bounds__(128)
f1_kernel(const float* __restrict__ v, float2* __restrict__ f1){
  __shared__ float  xin[4][132];
  __shared__ float2 tab[128];
  int tid = threadIdx.x;
  { float s,c; __sincosf(PI2 * (float)tid * (1.0f/128.0f), &s, &c); tab[tid] = make_float2(c,s); }
  long line0 = (long)blockIdx.x * 4;                 // line = ((bc*10+t)*128 + h)
  const float4* src = (const float4*)(v + line0*128);
  float4 d = src[tid];
  int ll = tid >> 5, ww = (tid & 31) << 2;
  xin[ll][ww+0]=d.x; xin[ll][ww+1]=d.y; xin[ll][ww+2]=d.z; xin[ll][ww+3]=d.w;
  __syncthreads();
  if (tid < 68){
    int lsub = tid / 17, kw = tid % 17;
    long line = line0 + lsub;
    float re = 0.f, im = 0.f;
    for (int w = 0; w < 128; ++w){
      float x = xin[lsub][w];
      float2 tw = tab[(kw*w) & 127];
      re += x * tw.x;
      im -= x * tw.y;
    }
    const float sc = 1.0f / 163840.0f;               // forward norm 1/(T*H*W)
    long h = line & 127, bct = line >> 7;
    f1[(bct*17 + kw)*128 + h] = make_float2(re*sc, im*sc);
  }
}

// F2: DFT over H, keep kh in {0..15,112..127}.  f1 -> f2 (B,C,KH32,KW17,T) complex
__global__ void __launch_bounds__(128)
f2_kernel(const float2* __restrict__ f1, float2* __restrict__ f2){
  __shared__ float2 lin[2][128];
  __shared__ float2 tab[128];
  int tid = threadIdx.x;
  { float s,c; __sincosf(PI2 * (float)tid * (1.0f/128.0f), &s, &c); tab[tid] = make_float2(c,s); }
  long line0 = (long)blockIdx.x * 2;                 // line = (bc*10+t)*17 + kw
  lin[0][tid] = f1[(line0+0)*128 + tid];
  lin[1][tid] = f1[(line0+1)*128 + tid];
  __syncthreads();
  if (tid < 64){
    int lsub = tid >> 5, khidx = tid & 31;
    long line = line0 + lsub;
    int kh = (khidx < 16) ? khidx : (96 + khidx);
    float re = 0.f, im = 0.f;
    for (int h = 0; h < 128; ++h){
      float2 a  = lin[lsub][h];
      float2 tw = tab[(kh*h) & 127];                 // e^{-i th}: (c,-s)
      re += a.x*tw.x + a.y*tw.y;
      im += a.y*tw.x - a.x*tw.y;
    }
    long kw = line % 17, bct = line / 17;
    long tt = bct % 10, bc = bct / 10;
    f2[((bc*32 + khidx)*17 + kw)*10 + tt] = make_float2(re, im);
  }
}

// F3: DFT over T(10), keep kt in {0..3,6..9}.  f2 -> f3 (B,C,MODES) complex
__global__ void __launch_bounds__(256)
f3_kernel(const float2* __restrict__ f2, float2* __restrict__ f3){
  long idx = (long)blockIdx.x*256 + threadIdx.x;     // 1,114,112
  long mode = idx % MODES, bc = idx / MODES;
  int ktidx = (int)(mode / 544);
  int rem   = (int)(mode % 544);
  int khidx = rem / 17, kw = rem % 17;
  int kt = (ktidx < 4) ? ktidx : (ktidx + 2);
  const float2* src = f2 + ((bc*32 + khidx)*17 + kw)*10;
  float re = 0.f, im = 0.f;
  #pragma unroll
  for (int tt = 0; tt < 10; ++tt){
    float2 a = src[tt];
    float s,c; __sincosf(PI2 * (float)((kt*tt)%10) * 0.1f, &s, &c);
    re += a.x*c + a.y*s;
    im += a.y*c - a.x*s;
  }
  f3[idx] = make_float2(re, im);
}

// Mode mix: out[b,o,m] = sum_i in[b,i,m] * (wr+i*wi)[i,o,m]; all 4 batches per thread
__global__ void __launch_bounds__(256)
mix_kernel(const float2* __restrict__ f3, const float* __restrict__ wr,
           const float* __restrict__ wi, float2* __restrict__ fm){
  long idx = (long)blockIdx.x*256 + threadIdx.x;     // 64*4352 = 278,528
  long mode = idx % MODES;
  int  o    = (int)(idx / MODES);
  float2 acc[4] = {{0.f,0.f},{0.f,0.f},{0.f,0.f},{0.f,0.f}};
  for (int i = 0; i < 64; ++i){
    long wofs = ((long)(i*64 + o))*MODES + mode;
    if (i + 4 < 64){                                  // global_prefetch_b8 ahead
      __builtin_prefetch(&wr[wofs + 4*64*MODES], 0, 1);
      __builtin_prefetch(&wi[wofs + 4*64*MODES], 0, 1);
    }
    float wre = wr[wofs], wim = wi[wofs];
    #pragma unroll
    for (int bb = 0; bb < 4; ++bb){
      float2 a = f3[((long)(bb*64 + i))*MODES + mode];
      acc[bb].x += a.x*wre - a.y*wim;
      acc[bb].y += a.x*wim + a.y*wre;
    }
  }
  #pragma unroll
  for (int bb = 0; bb < 4; ++bb)
    fm[((long)(bb*64 + o))*MODES + mode] = acc[bb];
}

// I1: inverse DFT over T.  fm -> i1 (B,O,KH32,KW17,T10) complex
__global__ void __launch_bounds__(256)
i1_kernel(const float2* __restrict__ fm, float2* __restrict__ i1){
  long idx = (long)blockIdx.x*256 + threadIdx.x;     // 1,392,640
  int  tt = (int)(idx % 10); long r = idx / 10;
  int  kw = (int)(r % 17);   r /= 17;
  int  khidx = (int)(r % 32); long bo = r / 32;
  float re = 0.f, im = 0.f;
  #pragma unroll
  for (int ktidx = 0; ktidx < 8; ++ktidx){
    int kt = (ktidx < 4) ? ktidx : (ktidx + 2);
    float2 a = fm[bo*MODES + (ktidx*32 + khidx)*17 + kw];
    float s,c; __sincosf(PI2 * (float)((kt*tt)%10) * 0.1f, &s, &c);
    re += a.x*c - a.y*s;                             // e^{+i th}
    im += a.x*s + a.y*c;
  }
  i1[idx] = make_float2(re, im);
}

// I2: inverse DFT over H (32 kept modes -> 128 h).  i1 -> i2 (B,O,T,H,KW17) complex
__global__ void __launch_bounds__(256)
i2_kernel(const float2* __restrict__ i1, float2* __restrict__ i2){
  __shared__ float2 tab[128];
  int tid = threadIdx.x;
  if (tid < 128){ float s,c; __sincosf(PI2 * (float)tid * (1.0f/128.0f), &s, &c); tab[tid] = make_float2(c,s); }
  __syncthreads();
  long idx = (long)blockIdx.x*256 + tid;             // 5,570,560
  int kw = (int)(idx % 17); long r = idx / 17;
  int h  = (int)(r % 128);  r /= 128;
  int tt = (int)(r % 10);   long bo = r / 10;
  float re = 0.f, im = 0.f;
  #pragma unroll
  for (int khidx = 0; khidx < 32; ++khidx){
    int kh = (khidx < 16) ? khidx : (96 + khidx);
    float2 a  = i1[((bo*32 + khidx)*17 + kw)*10 + tt];
    float2 tw = tab[(kh*h) & 127];                   // e^{+i th}
    re += a.x*tw.x - a.y*tw.y;
    im += a.x*tw.y + a.y*tw.x;
  }
  i2[idx] = make_float2(re, im);
}

// I3: hermitian inverse over W (17 modes -> 128 real).  i2 -> S (B,O,T,H,W) real
__global__ void __launch_bounds__(256)
i3_kernel(const float2* __restrict__ i2, float* __restrict__ S){
  __shared__ float2 xl[2][17];
  __shared__ float2 tab[128];
  int tid = threadIdx.x;
  if (tid < 128){ float s,c; __sincosf(PI2 * (float)tid * (1.0f/128.0f), &s, &c); tab[tid] = make_float2(c,s); }
  long line0 = (long)blockIdx.x * 2;                 // line = (bo*10+t)*128 + h
  if (tid < 34){
    int ll = tid / 17, kk = tid % 17;
    xl[ll][kk] = i2[(line0+ll)*17 + kk];
  }
  __syncthreads();
  int ll = tid >> 7, w = tid & 127;
  float acc = xl[ll][0].x;
  #pragma unroll
  for (int kw = 1; kw < 17; ++kw){
    float2 X  = xl[ll][kw];
    float2 tw = tab[(kw*w) & 127];
    acc += 2.0f*(X.x*tw.x - X.y*tw.y);
  }
  S[(line0+ll)*128 + w] = acc;
}

// ------------------------------- driver -------------------------------------------
extern "C" void kernel_launch(void* const* d_in, const int* in_sizes, int n_in,
                              void* d_out, int out_size, void* d_ws, size_t ws_size,
                              hipStream_t stream){
  (void)in_sizes; (void)n_in; (void)out_size; (void)ws_size;
  const float* x   = (const float*)d_in[0];
  const float* lw1 = (const float*)d_in[1];
  const float* lb1 = (const float*)d_in[2];
  const float* lw2 = (const float*)d_in[3];
  const float* lb2 = (const float*)d_in[4];
  const float* swr = (const float*)d_in[5];
  const float* swi = (const float*)d_in[6];
  const float* skw = (const float*)d_in[7];
  const float* skb = (const float*)d_in[8];
  const float* pw1 = (const float*)d_in[9];
  const float* pb1 = (const float*)d_in[10];
  const float* pw2 = (const float*)d_in[11];
  const float* pb2 = (const float*)d_in[12];
  float* out = (float*)d_out;

  char* ws = (char*)d_ws;
  size_t o = 0;
  float*  v  = (float*)(ws + o);  o += (size_t)4*CH*VOL*4;            // 167.8 MB
  float*  v2 = (float*)(ws + o);  o += (size_t)4*CH*VOL*4;            // 167.8 MB
  float*  Sb = (float*)(ws + o);  o += (size_t)4*CH*VOL*4;            // 167.8 MB
  float2* f1 = (float2*)(ws + o); o += (size_t)4*CH*10*17*128*8;      // 44.6 MB (also i2)
  float2* f2 = (float2*)(ws + o); o += (size_t)4*CH*32*17*10*8;       // 11.1 MB (also i1)
  float2* f3 = (float2*)(ws + o); o += (size_t)4*CH*MODES*8;          // 8.9 MB
  float2* fm = (float2*)(ws + o); o += (size_t)4*CH*MODES*8;          // 8.9 MB
  // fragment-packed weights (bf16): [tile-kstep][lane][16]
  bf16_t* pl1 = (bf16_t*)(ws + o); o += (size_t)16*32*16*2;           // lift1 M256 K8   (KS=1)
  bf16_t* pl2 = (bf16_t*)(ws + o); o += (size_t)32*32*16*2;           // lift2 M64  K256 (KS=8)
  bf16_t* psk = (bf16_t*)(ws + o); o += (size_t)4*8*32*16*2;          // skip  M64  K64  x4 layers
  bf16_t* pp1 = (bf16_t*)(ws + o); o += (size_t)32*32*16*2;           // proj1 M256 K64  (KS=2)
  bf16_t* pp2 = (bf16_t*)(ws + o); o += (size_t)8*32*16*2;            // proj2 M8   K256 (KS=8)

  // ---- cold: pack all weights into WMMA A-fragment order (bf16) ----
  repack_a_kernel<<<2, 256, 0, stream>>>(lw1, 256,   8, pl1, 16*32);
  repack_a_kernel<<<4, 256, 0, stream>>>(lw2,  64, 256, pl2, 32*32);
  for (int layer = 0; layer < 4; ++layer)
    repack_a_kernel<<<1, 256, 0, stream>>>(skw + (size_t)layer*CH*CH, 64, 64,
                                           psk + (size_t)layer*8*32*16, 8*32);
  repack_a_kernel<<<4, 256, 0, stream>>>(pw1, 256,  64, pp1, 32*32);
  repack_a_kernel<<<1, 256, 0, stream>>>(pw2,   8, 256, pp2, 8*32);

  copy_head_kernel<<<7680, 256, 0, stream>>>((const float4*)x, (float4*)out);
  mlp_wmma_kernel<0><<<10240, 256, 0, stream>>>(x, (const v16bf*)pl1, lb1,
                                                (const v16bf*)pl2, lb2, v);

  float* cur = v;
  float* nxt = v2;
  for (int layer = 0; layer < 4; ++layer){
    f1_kernel<<<81920, 128, 0, stream>>>(cur, f1);
    f2_kernel<<<21760, 128, 0, stream>>>(f1, f2);
    f3_kernel<<<4352, 256, 0, stream>>>(f2, f3);
    mix_kernel<<<1088, 256, 0, stream>>>(f3,
        swr + (size_t)layer*CH*CH*MODES, swi + (size_t)layer*CH*CH*MODES, fm);
    i1_kernel<<<5440, 256, 0, stream>>>(fm, f2);      // reuse f2 as i1
    i2_kernel<<<21760, 256, 0, stream>>>(f2, f1);     // reuse f1 as i2
    i3_kernel<<<163840, 256, 0, stream>>>(f1, Sb);
    skip_wmma_kernel<<<10240, 256, 0, stream>>>(cur, Sb,
        (const v16bf*)(psk + (size_t)layer*8*32*16), skb + (size_t)layer*CH,
        nxt, (layer < 3) ? 1 : 0);
    float* tmp = cur; cur = nxt; nxt = tmp;
  }

  mlp_wmma_kernel<1><<<10240, 256, 0, stream>>>(cur, (const v16bf*)pp1, pb1,
                                                (const v16bf*)pp2, pb2, out);
}
